// NetPerActionDynaQModel_82789789598536
// MI455X (gfx1250) — matrix operand order repaired and checked by reading, not compiled, not run on hardware
//
#include <hip/hip_runtime.h>
#include <hip/hip_bf16.h>

// Problem constants (match reference)
#define BB 16384
#define DD 512
#define HH 1024
#define AA 18
#define KSTEP 32   // WMMA K per step (bf16 16x16x32)

typedef __attribute__((ext_vector_type(16))) __bf16 v16bf;
typedef __attribute__((ext_vector_type(8)))  float  v8f;

union ABf { uint4 q[2]; v16bf v; };

static __device__ __forceinline__ unsigned short f2bf(float f) {
    union { float f; unsigned int u; } v; v.f = f;
    unsigned int u = v.u;
    u = (u + 0x7FFFu + ((u >> 16) & 1u)) >> 16;   // round-to-nearest-even
    return (unsigned short)u;
}

static __device__ __forceinline__ float bf2f(unsigned short h) {
    union { unsigned int u; float f; } v; v.u = ((unsigned int)h) << 16;
    return v.f;
}

// gfx1250 async copy: one 16B chunk global -> LDS per lane (ASYNCcnt-tracked)
static __device__ __forceinline__ void async_g2l_b128(unsigned int lds_off,
                                                      unsigned long long gaddr) {
    asm volatile("global_load_async_to_lds_b128 %0, %1, off"
                 :: "v"(lds_off), "v"(gaddr) : "memory");
}

// ---------------------------------------------------------------- prep kernels

__global__ void k_zero_counts(int* counts) {
    if (threadIdx.x < AA) counts[threadIdx.x] = 0;
}

__global__ void k_f2bf(const float* __restrict__ src, unsigned short* __restrict__ dst, int n) {
    int i = blockIdx.x * 256 + threadIdx.x;
    if (i < n) dst[i] = f2bf(src[i]);
}

// src [R,C] fp32 -> dst [C,R] bf16
__global__ void k_transpose_bf(const float* __restrict__ src, unsigned short* __restrict__ dst,
                               int R, int C) {
    int i = blockIdx.x * 256 + threadIdx.x;
    if (i >= R * C) return;
    int r = i / C, c = i % C;
    dst[(size_t)c * R + r] = f2bf(src[i]);
}

// obs_head_W [A,H,D] fp32 -> [A,D,H] bf16
__global__ void k_head_transpose(const float* __restrict__ src, unsigned short* __restrict__ dst) {
    int i = blockIdx.x * 256 + threadIdx.x;
    if (i >= AA * HH * DD) return;
    int a = i / (HH * DD);
    int r = i % (HH * DD);
    int h = r / DD, d = r % DD;
    dst[(size_t)a * DD * HH + (size_t)d * HH + h] = f2bf(src[i]);
}

// bin batch rows by action
__global__ void k_bin(const int* __restrict__ actions, int* counts, int* bins) {
    int b = blockIdx.x * 256 + threadIdx.x;
    if (b >= BB) return;
    int a = actions[b];
    int pos = atomicAdd(&counts[a], 1);
    bins[(size_t)a * BB + pos] = b;
}

// ---------------------------------------------------------------- encoder GEMMs
// f_enc = relu(obs @ W + b). Block tile: 128 rows x 64 cols, 8 waves (16 rows each).
// B tile (64n x 32k bf16 = 4KB) staged to LDS via async-to-LDS, double-buffered;
// all 8 waves share it. A fragments loaded straight from global (wave-unique rows).

__global__ __launch_bounds__(256) void k_encoders(
    const unsigned short* __restrict__ obs_bf, const unsigned short* __restrict__ wT,
    const float* __restrict__ b_obs, const float* __restrict__ b_rew,
    const float* __restrict__ b_done,
    unsigned short* __restrict__ f_obs, unsigned short* __restrict__ f_rew,
    unsigned short* __restrict__ f_done) {
    __shared__ unsigned short ldsB[2][64 * KSTEP];   // 2 x 4KB

    const int t = threadIdx.x;
    const int lane = t & 31;
    const int wv = t >> 5;

    const int blocksPerEnc = (BB / 128) * (HH / 64);   // 128*16 = 2048
    const int enc = blockIdx.x / blocksPerEnc;
    const int rem = blockIdx.x % blocksPerEnc;
    const int mb = rem / (HH / 64);
    const int ng = rem % (HH / 64);
    const int n0 = ng * 64;

    const unsigned short* Wt = wT + (size_t)enc * HH * DD;
    const float* bias = (enc == 0) ? b_obs : (enc == 1 ? b_rew : b_done);
    unsigned short* fout = (enc == 0) ? f_obs : (enc == 1 ? f_rew : f_done);

    // per-thread async-copy slot: thread t fills 16B of the 4KB tile
    const int cn = t >> 2;              // 0..63 (n within tile)
    const int cc = (t & 3) * 8;         // 0/8/16/24 (ushort chunk within k-row)
    const unsigned short* gsrc = Wt + (size_t)(n0 + cn) * DD + cc;
    const unsigned int lds0 = (unsigned int)(size_t)&ldsB[0][cn * KSTEP + cc];
    const unsigned int lds1 = (unsigned int)(size_t)&ldsB[1][cn * KSTEP + cc];

    const int row = mb * 128 + wv * 16 + (lane & 15);
    const unsigned short* Arow = obs_bf + (size_t)row * DD + ((lane >> 4) << 3);
    const int kb = (lane >> 4) << 3;

    async_g2l_b128(lds0, (unsigned long long)(size_t)gsrc);

    v8f acc[4] = {};
    const int NK = DD / KSTEP;          // 16
    for (int ks = 0; ks < NK; ++ks) {
        const int cur = ks & 1;

        // A-fragment global loads in flight across the sync point
        ABf afr;
        afr.q[0] = *(const uint4*)(Arow + ks * KSTEP);
        afr.q[1] = *(const uint4*)(Arow + ks * KSTEP + 16);

        if (ks + 1 < NK) {
            async_g2l_b128(cur ? lds0 : lds1,
                           (unsigned long long)(size_t)(gsrc + (ks + 1) * KSTEP));
            asm volatile("s_wait_asynccnt 0x1" ::: "memory");
        } else {
            asm volatile("s_wait_asynccnt 0x0" ::: "memory");
        }
        __syncthreads();                 // tile[cur] ready for everyone

        const unsigned short* Bl = ldsB[cur];
        ABf bfr[4];
#pragma unroll
        for (int j = 0; j < 4; ++j) {    // batch all ds_load_b128 first
            const int n = j * 16 + (lane & 15);
            bfr[j].q[0] = *(const uint4*)(Bl + n * KSTEP + kb);
            bfr[j].q[1] = *(const uint4*)(Bl + n * KSTEP + kb + 16);
        }
#pragma unroll
        for (int j = 0; j < 4; ++j) {    // then WMMAs back-to-back
            acc[j] = __builtin_amdgcn_wmma_f32_16x16x32_bf16(
                false, afr.v, false, bfr[j].v, (short)0, acc[j], false, false);
        }
        __syncthreads();                 // everyone done reading tile[cur]
    }

    const int mrow0 = mb * 128 + wv * 16 + ((lane >> 4) << 3);
#pragma unroll
    for (int j = 0; j < 4; ++j) {
        const int n = n0 + j * 16 + (lane & 15);
        const float bv = bias[n];
#pragma unroll
        for (int r = 0; r < 8; ++r) {
            float v = acc[j][r] + bv;
            v = v > 0.0f ? v : 0.0f;
            fout[(size_t)(mrow0 + r) * HH + n] = f2bf(v);
        }
    }
}

// ---------------------------------------------------------------- rew/done heads

__global__ __launch_bounds__(256) void k_rew_done(
    const unsigned short* __restrict__ f_rew, const unsigned short* __restrict__ f_done,
    const float* __restrict__ rw, const float* __restrict__ rb,
    const float* __restrict__ dw, const float* __restrict__ db,
    const int* __restrict__ actions, float* __restrict__ out_rew,
    float* __restrict__ out_done) {
    const int wv = threadIdx.x >> 5;
    const int lane = threadIdx.x & 31;
    const int b = blockIdx.x * 8 + wv;
    if (b >= BB) return;
    const int a = actions[b];
    float sr = 0.0f, sd = 0.0f;
    for (int h = lane; h < HH; h += 32) {
        sr += bf2f(f_rew[(size_t)b * HH + h]) * rw[a * HH + h];
        sd += bf2f(f_done[(size_t)b * HH + h]) * dw[a * HH + h];
    }
#pragma unroll
    for (int off = 16; off; off >>= 1) {
        sr += __shfl_xor(sr, off, 32);
        sd += __shfl_xor(sd, off, 32);
    }
    if (lane == 0) {
        out_rew[b] = sr + rb[a];
        float z = sd + db[a];
        out_done[b] = 1.0f / (1.0f + __expf(-z));
    }
}

// ---------------------------------------------------------------- obs head GEMM
// Block = (action a, 128 gathered rows, 64 of D's columns). Same async-LDS
// double-buffered pipeline as the encoders; A rows gathered through bins[].

__global__ __launch_bounds__(256) void k_obs_head(
    const unsigned short* __restrict__ f_obs, const unsigned short* __restrict__ headWT,
    const float* __restrict__ head_b, const float* __restrict__ obs,
    const int* __restrict__ counts, const int* __restrict__ bins,
    float* __restrict__ out) {
    const int blocksPerA = (BB / 128) * (DD / 64);   // 128*8 = 1024
    const int a = blockIdx.x / blocksPerA;
    const int rem = blockIdx.x % blocksPerA;
    const int mb = rem / (DD / 64);
    const int ngB = rem % (DD / 64);
    const int n0 = ngB * 64;

    const int cnt = counts[a];
    if (mb * 128 >= cnt) return;        // uniform per block

    __shared__ unsigned short ldsB[2][64 * KSTEP];   // 2 x 4KB
    __shared__ int rows[128];
    if (threadIdx.x < 128) {
        int idx = mb * 128 + (int)threadIdx.x;
        rows[threadIdx.x] = bins[(size_t)a * BB + (idx < cnt ? idx : cnt - 1)];
    }
    __syncthreads();

    const int t = threadIdx.x;
    const int lane = t & 31;
    const int wv = t >> 5;

    const unsigned short* Wt = headWT + (size_t)a * DD * HH;
    const int cn = t >> 2;
    const int cc = (t & 3) * 8;
    const unsigned short* gsrc = Wt + (size_t)(n0 + cn) * HH + cc;
    const unsigned int lds0 = (unsigned int)(size_t)&ldsB[0][cn * KSTEP + cc];
    const unsigned int lds1 = (unsigned int)(size_t)&ldsB[1][cn * KSTEP + cc];

    const int arow = rows[wv * 16 + (lane & 15)];
    const unsigned short* Arow = f_obs + (size_t)arow * HH + ((lane >> 4) << 3);
    const int kb = (lane >> 4) << 3;

    async_g2l_b128(lds0, (unsigned long long)(size_t)gsrc);

    v8f acc[4] = {};
    const int NK = HH / KSTEP;          // 32
    for (int ks = 0; ks < NK; ++ks) {
        const int cur = ks & 1;

        ABf afr;
        afr.q[0] = *(const uint4*)(Arow + ks * KSTEP);
        afr.q[1] = *(const uint4*)(Arow + ks * KSTEP + 16);

        if (ks + 1 < NK) {
            async_g2l_b128(cur ? lds0 : lds1,
                           (unsigned long long)(size_t)(gsrc + (ks + 1) * KSTEP));
            asm volatile("s_wait_asynccnt 0x1" ::: "memory");
        } else {
            asm volatile("s_wait_asynccnt 0x0" ::: "memory");
        }
        __syncthreads();

        const unsigned short* Bl = ldsB[cur];
        ABf bfr[4];
#pragma unroll
        for (int j = 0; j < 4; ++j) {
            const int n = j * 16 + (lane & 15);
            bfr[j].q[0] = *(const uint4*)(Bl + n * KSTEP + kb);
            bfr[j].q[1] = *(const uint4*)(Bl + n * KSTEP + kb + 16);
        }
#pragma unroll
        for (int j = 0; j < 4; ++j) {
            acc[j] = __builtin_amdgcn_wmma_f32_16x16x32_bf16(
                false, afr.v, false, bfr[j].v, (short)0, acc[j], false, false);
        }
        __syncthreads();
    }

    const int hi = lane >> 4;
#pragma unroll
    for (int j = 0; j < 4; ++j) {
        const int n = n0 + j * 16 + (lane & 15);
        const float bv = head_b[a * DD + n];
#pragma unroll
        for (int r = 0; r < 8; ++r) {
            const int M = wv * 16 + r + 8 * hi;
            const int gi = mb * 128 + M;
            if (gi < cnt) {
                const int b = rows[M];
                out[(size_t)b * DD + n] = acc[j][r] + bv + obs[(size_t)b * DD + n];
            }
        }
    }
}

// ---------------------------------------------------------------- launch

extern "C" void kernel_launch(void* const* d_in, const int* in_sizes, int n_in,
                              void* d_out, int out_size, void* d_ws, size_t ws_size,
                              hipStream_t stream) {
    (void)in_sizes; (void)n_in; (void)out_size; (void)ws_size;

    const float* obs        = (const float*)d_in[0];
    const int*   actions    = (const int*)d_in[1];
    const float* obs_enc_W  = (const float*)d_in[2];
    const float* obs_enc_b  = (const float*)d_in[3];
    const float* rew_enc_W  = (const float*)d_in[4];
    const float* rew_enc_b  = (const float*)d_in[5];
    const float* done_enc_W = (const float*)d_in[6];
    const float* done_enc_b = (const float*)d_in[7];
    const float* obs_head_W = (const float*)d_in[8];
    const float* obs_head_b = (const float*)d_in[9];
    const float* rew_head_W = (const float*)d_in[10];
    const float* rew_head_b = (const float*)d_in[11];
    const float* done_head_W= (const float*)d_in[12];
    const float* done_head_b= (const float*)d_in[13];

    unsigned short* ws16   = (unsigned short*)d_ws;
    unsigned short* obs_bf = ws16;                                 // B*D
    unsigned short* wT     = obs_bf + (size_t)BB * DD;             // 3*H*D
    unsigned short* headWT = wT + (size_t)3 * HH * DD;             // A*D*H
    unsigned short* f_obs  = headWT + (size_t)AA * DD * HH;        // B*H
    unsigned short* f_rew  = f_obs + (size_t)BB * HH;              // B*H
    unsigned short* f_done = f_rew + (size_t)BB * HH;              // B*H
    int* counts = (int*)(f_done + (size_t)BB * HH);                // A (padded to 32)
    int* bins   = counts + 32;                                     // A*B

    float* out_obs  = (float*)d_out;
    float* out_rew  = out_obs + (size_t)BB * DD;
    float* out_done = out_rew + BB;

    k_zero_counts<<<1, 32, 0, stream>>>(counts);
    k_f2bf<<<(BB * DD + 255) / 256, 256, 0, stream>>>(obs, obs_bf, BB * DD);
    k_transpose_bf<<<(DD * HH + 255) / 256, 256, 0, stream>>>(obs_enc_W,  wT,                       DD, HH);
    k_transpose_bf<<<(DD * HH + 255) / 256, 256, 0, stream>>>(rew_enc_W,  wT + (size_t)HH * DD,     DD, HH);
    k_transpose_bf<<<(DD * HH + 255) / 256, 256, 0, stream>>>(done_enc_W, wT + (size_t)2 * HH * DD, DD, HH);
    k_head_transpose<<<(AA * HH * DD + 255) / 256, 256, 0, stream>>>(obs_head_W, headWT);
    k_bin<<<(BB + 255) / 256, 256, 0, stream>>>(actions, counts, bins);

    // 3 enc * 128 row-blocks * 16 col-blocks = 6144 blocks
    k_encoders<<<6144, 256, 0, stream>>>(obs_bf, wT, obs_enc_b, rew_enc_b, done_enc_b,
                                         f_obs, f_rew, f_done);

    k_rew_done<<<BB / 8, 256, 0, stream>>>(f_rew, f_done, rew_head_W, rew_head_b,
                                           done_head_W, done_head_b, actions,
                                           out_rew, out_done);

    // 18 actions * 128 row-block capacity * 8 col-blocks
    k_obs_head<<<AA * (BB / 128) * (DD / 64), 256, 0, stream>>>(
        f_obs, headWT, obs_head_b, obs, counts, bins, out_obs);
}
